// EdgeAttention_33397665694635
// MI455X (gfx1250) — matrix operand order repaired and checked by reading, not compiled, not run on hardware
//
#include <hip/hip_runtime.h>

// EdgeAttention for MI455X (gfx1250, wave32, WMMA + Tensor Data Mover).
// All matmuls in bf16 WMMA (v_wmma_f32_16x16x32_bf16), f32 accumulate.
// Stage 2 is a fused flash-style attention per (b,l) using ~280KB of WGP LDS,
// with the LDS tiles filled by TDM tensor_load_to_lds (hardware row padding).

#define BB 2
#define NN 256
#define DD 128
constexpr int M_TOTAL = BB * NN * NN;   // 131072 flat rows
constexpr int MT = 64;                  // M-tile for projection GEMMs

typedef __bf16 bf16_t;
typedef bf16_t v16bf __attribute__((ext_vector_type(16)));
typedef float  v8f   __attribute__((ext_vector_type(8)));

// ---- bf16 <-> f32 as raw bits (round-to-nearest-even) -----------------------
__device__ inline unsigned short f2bf(float f) {
    unsigned int u = __float_as_uint(f);
    u += 0x7FFFu + ((u >> 16) & 1u);
    return (unsigned short)(u >> 16);
}
__device__ inline float bf2f(unsigned short h) {
    return __uint_as_float(((unsigned int)h) << 16);
}

// ---- WMMA fragment loaders (layouts per CDNA5 ISA §7.12.2) -------
// A 16x32 bf16: lane(m=L&15,h=L>>4): elems 0..7 = row[k0+h*8 .. +7],
//                                    elems 8..15 = row[k0+16+h*8 .. +7]
__device__ inline v16bf load_a_frag(const unsigned short* lds, int pitch,
                                    int row0, int k0) {
    const int lane = threadIdx.x & 31;
    const int m = lane & 15, h = lane >> 4;
    const unsigned short* p = lds + (row0 + m) * pitch + k0 + h * 8;
    union { uint4 q[2]; v16bf v; } u;
    u.q[0] = *(const uint4*)p;
    u.q[1] = *(const uint4*)(p + 16);
    return u.v;
}
// B 32x16 bf16: lane(n=L&15,h=L>>4): elems 0..7 = K h*8 .. +7,
// elems 8..15 = K 16+h*8 .. +7 of column n, sourced from row (n0+n).
__device__ inline v16bf load_b_frag(const unsigned short* lds, int pitch,
                                    int n0, int c0) {
    const int lane = threadIdx.x & 31;
    const int n = lane & 15, h = lane >> 4;
    const unsigned short* p = lds + (n0 + n) * pitch + c0 + h * 8;
    union { uint4 q[2]; v16bf v; } u;
    u.q[0] = *(const uint4*)p;
    u.q[1] = *(const uint4*)(p + 16);
    return u.v;
}
__device__ inline v8f wmma_bf16(v16bf a, v16bf b, v8f c) {
    return __builtin_amdgcn_wmma_f32_16x16x32_bf16(
        false, a, false, b, (short)0, c, false, false);
}

// ---- cross-lane row reductions within a 16-lane half ------------------------
__device__ inline float rowred_max(float v) {
    v = fmaxf(v, __shfl_xor(v, 1, 32));
    v = fmaxf(v, __shfl_xor(v, 2, 32));
    v = fmaxf(v, __shfl_xor(v, 4, 32));
    v = fmaxf(v, __shfl_xor(v, 8, 32));
    return v;
}
__device__ inline float rowred_sum(float v) {
    v += __shfl_xor(v, 1, 32);
    v += __shfl_xor(v, 2, 32);
    v += __shfl_xor(v, 4, 32);
    v += __shfl_xor(v, 8, 32);
    return v;
}

// ---- Tensor Data Mover (CDNA5 ISA) --------------------
#if __has_builtin(__builtin_amdgcn_tensor_load_to_lds)
#define HAVE_TDM 1
typedef unsigned int u32x4 __attribute__((ext_vector_type(4)));
typedef int          i32x4 __attribute__((ext_vector_type(4)));
typedef int          i32x8 __attribute__((ext_vector_type(8)));

__device__ inline unsigned lds_byte_off(const void* p) {
    // generic -> LDS(AS3) addrspacecast; AS3 pointers are 32-bit offsets
    return (unsigned int)(__attribute__((address_space(3))) const char*)p;
}
// D# group0: count=1 (valid user desc), lds_addr[63:32], global_addr[120:64],
// type=2 ("image") at [127:126].
__device__ inline u32x4 tdm_group0(unsigned lds_addr, unsigned long long ga) {
    u32x4 g;
    g[0] = 1u;
    g[1] = lds_addr;
    g[2] = (unsigned)(ga & 0xFFFFFFFFu);
    g[3] = ((unsigned)(ga >> 32) & 0x01FFFFFFu) | (2u << 30);
    return g;
}
// D# group1: 2D tile, data_size=4B (code 2), LDS pad enabled.
//   tile_w/tensor_w in u32 units, tile_h/tensor_h rows, stride_u32 row stride,
//   pad: pad_amt_dw DWORDs after every 2^(pad_iv_code+1) DWORDs stored.
__device__ inline i32x8 tdm_group1_2d(unsigned tile_w, unsigned tile_h,
                                      unsigned long long stride_u32,
                                      unsigned tensor_w, unsigned tensor_h,
                                      unsigned pad_iv_code, unsigned pad_amt_dw) {
    unsigned w0 = (2u << 16)                        // data_size = 4 bytes
                | (1u << 20)                        // pad_enable
                | (pad_iv_code << 22)               // pad_interval
                | ((pad_amt_dw - 1u) << 25);        // pad_amount code (0 => 1)
    unsigned w1 = (tensor_w & 0xFFFFu) << 16;                       // dim0[15:0]
    unsigned w2 = (tensor_w >> 16) | ((tensor_h & 0xFFFFu) << 16);  // dim0 hi | dim1 lo
    unsigned w3 = (tensor_h >> 16) | ((tile_w & 0xFFFFu) << 16);    // dim1 hi | tile0
    unsigned w4 = (tile_h & 0xFFFFu);                               // tile1 (tile2=0)
    unsigned w5 = (unsigned)(stride_u32 & 0xFFFFFFFFu);             // stride0[31:0]
    unsigned w6 = (unsigned)(stride_u32 >> 32) & 0xFFFFu;           // stride0[47:32]
    unsigned w7 = 0;                                                // stride1 unused (2D)
    i32x8 g;
    g[0]=(int)w0; g[1]=(int)w1; g[2]=(int)w2; g[3]=(int)w3;
    g[4]=(int)w4; g[5]=(int)w5; g[6]=(int)w6; g[7]=(int)w7;
    return g;
}
__device__ inline void tdm_load_2d(const void* gptr, const void* ldsptr,
                                   unsigned tile_w, unsigned tile_h,
                                   unsigned long long stride_u32,
                                   unsigned pad_iv_code, unsigned pad_amt_dw) {
    u32x4 g0 = tdm_group0(lds_byte_off(ldsptr), (unsigned long long)(size_t)gptr);
    i32x8 g1 = tdm_group1_2d(tile_w, tile_h, stride_u32, tile_w, tile_h,
                             pad_iv_code, pad_amt_dw);
    i32x4 z4; z4[0]=0; z4[1]=0; z4[2]=0; z4[3]=0;
#if __has_include(<hip/amd_detail/amd_gfx1250_TDM.h>)
    // amdgpu-toolchain (clang-23 / therock-10.0): 6-arg form
    i32x8 z8; z8[0]=0; z8[1]=0; z8[2]=0; z8[3]=0; z8[4]=0; z8[5]=0; z8[6]=0; z8[7]=0;
    __builtin_amdgcn_tensor_load_to_lds(g0, g1, z4, z4, z8, 0);
#else
    // ROCm 7.2 (clang-22): 5-arg form
    __builtin_amdgcn_tensor_load_to_lds(g0, g1, z4, z4, 0);
#endif
}
#endif  // HAVE_TDM

// =============================================================================
// Kernel 1: projections.  y = x @ W^T + b for Wq,Wk,Wv1,Wv2 in one pass.
// q,k,v1 stored bf16 row-major; v2 stored TRANSPOSED per (b,l): v2t[b][l][d][j]
// (C-tile lanes hold 8 consecutive rows at fixed column -> one b128 store).
// =============================================================================
constexpr int PA = 136;  // LDS pitch (elements), padded for bank rotation
constexpr int SMEM_PROJ = (MT * PA + 4 * DD * PA) * 2 + 4 * DD * 4;

__global__ __launch_bounds__(256) void proj_kernel(
    const float* __restrict__ ef,
    const float* __restrict__ Wq, const float* __restrict__ bq,
    const float* __restrict__ Wk, const float* __restrict__ bk,
    const float* __restrict__ Wv1, const float* __restrict__ bv1,
    const float* __restrict__ Wv2, const float* __restrict__ bv2,
    unsigned short* __restrict__ q_out, unsigned short* __restrict__ k_out,
    unsigned short* __restrict__ v1_out, unsigned short* __restrict__ v2t_out)
{
    extern __shared__ __align__(16) unsigned char smem[];
    unsigned short* xs  = (unsigned short*)smem;      // 64 x PA
    unsigned short* wsm = xs + MT * PA;               // 4 x 128 x PA
    float*          bias = (float*)(wsm + 4 * DD * PA);

    const int tid = threadIdx.x;
    const size_t m0 = (size_t)blockIdx.x * MT;

    for (int idx = tid; idx < MT * DD; idx += 256) {
        int r = idx >> 7, c = idx & 127;
        xs[r * PA + c] = f2bf(ef[(m0 + r) * DD + c]);
    }
    for (int idx = tid; idx < DD * DD; idx += 256) {
        int r = idx >> 7, c = idx & 127;
        int o = r * PA + c;
        wsm[0 * DD * PA + o] = f2bf(Wq[idx]);
        wsm[1 * DD * PA + o] = f2bf(Wk[idx]);
        wsm[2 * DD * PA + o] = f2bf(Wv1[idx]);
        wsm[3 * DD * PA + o] = f2bf(Wv2[idx]);
    }
    if (tid < DD) {
        bias[0 * DD + tid] = bq[tid];
        bias[1 * DD + tid] = bk[tid];
        bias[2 * DD + tid] = bv1[tid];
        bias[3 * DD + tid] = bv2[tid];
    }
    __syncthreads();

    const int wave = tid >> 5, lane = tid & 31;
    const int n = lane & 15, half = lane >> 4;
    const int n0 = wave * 16;  // 8 waves cover 128 output columns

    for (int wt = 0; wt < 4; ++wt) {
        for (int mg = 0; mg < 4; ++mg) {
            const float bc = bias[wt * DD + n0 + n];
            v8f acc;
            #pragma unroll
            for (int r = 0; r < 8; ++r) acc[r] = bc;
            #pragma unroll
            for (int kc = 0; kc < 4; ++kc) {
                v16bf a = load_a_frag(xs, PA, mg * 16, kc * 32);
                v16bf b = load_b_frag(wsm + wt * DD * PA, PA, n0, kc * 32);
                acc = wmma_bf16(a, b, acc);
            }
            const size_t mrow = m0 + (size_t)(mg * 16 + half * 8);
            if (wt == 3) {
                // transposed v2 store: one 16B store of 8 consecutive j
                size_t bi = mrow >> 16, rem = mrow & 65535;
                size_t l = rem >> 8, j0 = rem & 255;
                unsigned short tmp[8];
                #pragma unroll
                for (int r = 0; r < 8; ++r) tmp[r] = f2bf(acc[r]);
                size_t off = ((bi * NN + l) * DD + (size_t)(n0 + n)) * NN + j0;
                *(uint4*)(v2t_out + off) = *(const uint4*)tmp;
            } else {
                unsigned short* outp = (wt == 0) ? q_out : (wt == 1) ? k_out : v1_out;
                #pragma unroll
                for (int r = 0; r < 8; ++r)
                    outp[(mrow + r) * DD + n0 + n] = f2bf(acc[r]);
            }
        }
    }
}

// =============================================================================
// Kernel 2: fused attention per (b,l) with online softmax.
// LDS tiles filled by TDM (hardware-inserted row padding) when available.
// =============================================================================
constexpr int PB = 136;   // pitch for q/k/v1 tiles (256 x 128): 256B row + 16B pad
constexpr int PT = 264;   // pitch for v2^T tile (128 x 256):    512B row + 16B pad
constexpr int PS = 40;    // pitch for per-wave P staging (16 x 32)
constexpr int SMEM_ATTN = (3 * NN * PB + DD * PT + 8 * 16 * PS) * 2;  // 286720

__global__ __launch_bounds__(256) void attn_kernel(
    const unsigned short* __restrict__ qg, const unsigned short* __restrict__ kg,
    const unsigned short* __restrict__ v1g, const unsigned short* __restrict__ v2tg,
    unsigned short* __restrict__ ctxg)
{
    extern __shared__ __align__(16) unsigned char smem[];
    unsigned short* qs   = (unsigned short*)smem;     // 256 x PB
    unsigned short* ks   = qs + NN * PB;              // 256 x PB
    unsigned short* v1s  = ks + NN * PB;              // 256 x PB
    unsigned short* v2ts = v1s + NN * PB;             // 128 x PT
    unsigned short* pst  = v2ts + DD * PT;            // 8 waves x 16 x PS

    const int tid = threadIdx.x;
    const int bi = blockIdx.x >> 8;
    const int l  = blockIdx.x & 255;
    const int wave = tid >> 5, lane = tid & 31;

    const size_t qbase  = ((size_t)bi * NN * NN + l) * DD;            // +i*N*D+c
    const size_t kbase  = ((size_t)bi * NN * NN + (size_t)l * NN) * DD;
    const size_t v2base = (((size_t)bi * NN + l) * DD) * NN;          // +d*N+j

#ifdef HAVE_TDM
    if (wave == 0) {
        // q, v1: 2D tiles of 256 rows x 256B, global row stride 64KB.
        // Rows are 64 DWORDs; pad 4 DWORDs after every 64 (code 5) => pitch PB.
        tdm_load_2d(qg  + qbase, qs,  64u, 256u, 16384ull, 5u, 4u);
        tdm_load_2d(v1g + qbase, v1s, 64u, 256u, 16384ull, 5u, 4u);
        // k: contiguous 256 x 256B.
        tdm_load_2d(kg  + kbase, ks,  64u, 256u, 64ull,    5u, 4u);
        // v2^T: contiguous 128 x 512B; pad 4 DWORDs every 128 (code 6) => PT.
        tdm_load_2d(v2tg + v2base, v2ts, 128u, 128u, 128ull, 6u, 4u);
        __builtin_amdgcn_s_wait_tensorcnt(0);
    }
#else
    for (int idx = tid; idx < NN * DD; idx += 256) {
        int r = idx >> 7, c = idx & 127;
        qs [r * PB + c] = qg [qbase + (size_t)r * NN * DD + c];
        v1s[r * PB + c] = v1g[qbase + (size_t)r * NN * DD + c];
        ks [r * PB + c] = kg [kbase + idx];
    }
    for (int idx = tid; idx < DD * NN; idx += 256) {
        int r = idx >> 8, c = idx & 255;
        v2ts[r * PT + c] = v2tg[v2base + idx];
    }
#endif
    __syncthreads();

    const int n = lane & 15, half = lane >> 4;
    unsigned short* mypst = pst + wave * 16 * PS;
    const float scale = 0.08838834764831845f;  // 1/sqrt(128)
    const float NEG_INF = -__builtin_inff();

    for (int rep = 0; rep < 2; ++rep) {
        const int i0 = (wave + rep * 8) * 16;   // 16-row i-strip
        float mrun[8], lrun[8];
        v8f acc[8];
        #pragma unroll
        for (int r = 0; r < 8; ++r) { mrun[r] = NEG_INF; lrun[r] = 0.f; }
        #pragma unroll
        for (int dg = 0; dg < 8; ++dg)
            #pragma unroll
            for (int r = 0; r < 8; ++r) acc[dg][r] = 0.f;

        for (int jb = 0; jb < 8; ++jb) {
            const int j0 = jb * 32;
            v8f s0, s1;
            #pragma unroll
            for (int r = 0; r < 8; ++r) { s0[r] = 0.f; s1[r] = 0.f; }
            #pragma unroll
            for (int kc = 0; kc < 4; ++kc) {
                v16bf a  = load_a_frag(qs, PB, i0, kc * 32);
                v16bf b0 = load_b_frag(ks, PB, j0,      kc * 32);
                v16bf b1 = load_b_frag(ks, PB, j0 + 16, kc * 32);
                s0 = wmma_bf16(a, b0, s0);
                s1 = wmma_bf16(a, b1, s1);
            }
            // online softmax (rows m = r + 8*half; stats replicated over half)
            float p0[8], p1[8];
            #pragma unroll
            for (int r = 0; r < 8; ++r) {
                float a0 = s0[r] * scale, a1 = s1[r] * scale;
                float rm = rowred_max(fmaxf(a0, a1));
                float mn = fmaxf(mrun[r], rm);
                float corr = __expf(mrun[r] - mn);
                float e0 = __expf(a0 - mn), e1 = __expf(a1 - mn);
                lrun[r] = lrun[r] * corr + rowred_sum(e0 + e1);
                mrun[r] = mn;
                p0[r] = e0; p1[r] = e1;
                #pragma unroll
                for (int dg = 0; dg < 8; ++dg) acc[dg][r] *= corr;
            }
            // stage P (16x32 bf16) through per-wave LDS to re-layout C->A
            #pragma unroll
            for (int r = 0; r < 8; ++r) {
                int m = r + half * 8;
                mypst[m * PS + n]      = f2bf(p0[r]);
                mypst[m * PS + 16 + n] = f2bf(p1[r]);
            }
            asm volatile("s_wait_dscnt 0" ::: "memory");  // LDS write->read, same wave
            v16bf ap = load_a_frag(mypst, PS, 0, 0);
            #pragma unroll
            for (int dg = 0; dg < 8; ++dg) {
                v16bf bv = load_b_frag(v2ts, PT, dg * 16, j0);
                acc[dg] = wmma_bf16(ap, bv, acc[dg]);
            }
        }
        // epilogue: /= rowsum, *= v1, store ctx (bf16, row-major flat (b,i,l))
        float inv[8];
        #pragma unroll
        for (int r = 0; r < 8; ++r) inv[r] = 1.0f / lrun[r];
        #pragma unroll
        for (int dg = 0; dg < 8; ++dg) {
            #pragma unroll
            for (int r = 0; r < 8; ++r) {
                int i = i0 + r + half * 8;
                int c = dg * 16 + n;
                float val = acc[dg][r] * inv[r] * bf2f(v1s[i * PB + c]);
                ctxg[(((size_t)bi * NN + i) * NN + l) * DD + c] = f2bf(val);
            }
        }
    }
}

// =============================================================================
// Kernel 3: output projection  out = ctx @ Wo^T + bo   (f32 output)
// =============================================================================
constexpr int SMEM_OUT = (MT * PA + DD * PA) * 2 + DD * 4;

__global__ __launch_bounds__(256) void out_kernel(
    const unsigned short* __restrict__ ctxg,
    const float* __restrict__ Wo, const float* __restrict__ bo,
    float* __restrict__ out)
{
    extern __shared__ __align__(16) unsigned char smem[];
    unsigned short* xs  = (unsigned short*)smem;   // 64 x PA
    unsigned short* wsm = xs + MT * PA;            // 128 x PA
    float*          bias = (float*)(wsm + DD * PA);

    const int tid = threadIdx.x;
    const size_t m0 = (size_t)blockIdx.x * MT;

    for (int idx = tid; idx < MT * DD; idx += 256) {
        int r = idx >> 7, c = idx & 127;
        xs[r * PA + c] = ctxg[(m0 + r) * DD + c];
    }
    for (int idx = tid; idx < DD * DD; idx += 256) {
        int r = idx >> 7, c = idx & 127;
        wsm[r * PA + c] = f2bf(Wo[idx]);
    }
    if (tid < DD) bias[tid] = bo[tid];
    __syncthreads();

    const int wave = tid >> 5, lane = tid & 31;
    const int n = lane & 15, half = lane >> 4;
    const int n0 = wave * 16;

    for (int mg = 0; mg < 4; ++mg) {
        const float bc = bias[n0 + n];
        v8f acc;
        #pragma unroll
        for (int r = 0; r < 8; ++r) acc[r] = bc;
        #pragma unroll
        for (int kc = 0; kc < 4; ++kc) {
            v16bf a = load_a_frag(xs, PA, mg * 16, kc * 32);
            v16bf b = load_b_frag(wsm, PA, n0, kc * 32);
            acc = wmma_bf16(a, b, acc);
        }
        const size_t mrow = m0 + (size_t)(mg * 16 + half * 8);
        #pragma unroll
        for (int r = 0; r < 8; ++r)
            out[(mrow + r) * DD + n0 + n] = acc[r];
    }
}

// =============================================================================
extern "C" void kernel_launch(void* const* d_in, const int* in_sizes, int n_in,
                              void* d_out, int out_size, void* d_ws, size_t ws_size,
                              hipStream_t stream) {
    const float* ef  = (const float*)d_in[0];
    const float* Wq  = (const float*)d_in[1];
    const float* bq  = (const float*)d_in[2];
    const float* Wk  = (const float*)d_in[3];
    const float* bk  = (const float*)d_in[4];
    const float* Wv1 = (const float*)d_in[5];
    const float* bv1 = (const float*)d_in[6];
    const float* Wv2 = (const float*)d_in[7];
    const float* bv2 = (const float*)d_in[8];
    const float* Wo  = (const float*)d_in[9];
    const float* bo  = (const float*)d_in[10];
    float* out = (float*)d_out;

    // Workspace layout (bf16): q, k, v1, v2^T, ctx — 5 x 32MB = 160MB
    const size_t elems = (size_t)M_TOTAL * DD;
    unsigned short* q   = (unsigned short*)d_ws;
    unsigned short* k   = q   + elems;
    unsigned short* v1  = k   + elems;
    unsigned short* v2t = v1  + elems;
    unsigned short* ctx = v2t + elems;

    // Opt in to >64KB dynamic LDS (executes immediately; idempotent).
    static bool attr_done = false;
    if (!attr_done) {
        hipFuncSetAttribute((const void*)proj_kernel,
                            hipFuncAttributeMaxDynamicSharedMemorySize, SMEM_PROJ);
        hipFuncSetAttribute((const void*)attn_kernel,
                            hipFuncAttributeMaxDynamicSharedMemorySize, SMEM_ATTN);
        hipFuncSetAttribute((const void*)out_kernel,
                            hipFuncAttributeMaxDynamicSharedMemorySize, SMEM_OUT);
        attr_done = true;
    }

    dim3 blk(256);
    proj_kernel<<<M_TOTAL / MT, blk, SMEM_PROJ, stream>>>(
        ef, Wq, bq, Wk, bk, Wv1, bv1, Wv2, bv2, q, k, v1, v2t);
    attn_kernel<<<BB * NN, blk, SMEM_ATTN, stream>>>(q, k, v1, v2t, ctx);
    out_kernel<<<M_TOTAL / MT, blk, SMEM_OUT, stream>>>(ctx, Wo, bo, out);
}
